// SI_GNN_52424370815008
// MI455X (gfx1250) — matrix operand order repaired and checked by reading, not compiled, not run on hardware
//
#include <hip/hip_runtime.h>
#include <math.h>

// ---------------- CDNA5 vector types for WMMA ----------------
typedef __attribute__((ext_vector_type(16))) __bf16 v16bf;
typedef __attribute__((ext_vector_type(8)))  __bf16 v8bf;
typedef __attribute__((ext_vector_type(8)))  float  v8f;
typedef int v4i __attribute__((vector_size(16)));   // matches async-to-LDS builtin param

// ---------------- problem constants ----------------
#define B_    256
#define N_    50
#define M_    100
#define FSD_  300
#define FVD_  2048
#define LDIM_ 2048
#define BBD_  128
#define INTER_ 512
#define KH_   8
#define DV_   2176   // FVD + BBD
#define DSI_  428    // FSD + BBD
#define DSIP_ 448    // DSI padded to K-step multiple (32)
#define SOUT_ 600    // FSD + FSD

#define NEG_INF (-__builtin_inff())

// ---------------- CDNA5 async-to-LDS (probe-confirmed present) ----------------
#if __has_builtin(__builtin_amdgcn_global_load_async_to_lds_b128)
#define HAVE_ASYNC_LDS 1
#endif

__device__ inline void async_cp16(const __bf16* g, __bf16* l) {
#ifdef HAVE_ASYNC_LDS
    __builtin_amdgcn_global_load_async_to_lds_b128(
        (__attribute__((address_space(1))) v4i*)g,
        (__attribute__((address_space(3))) v4i*)l, 0, 0);
#else
    *(v8bf*)l = *(const v8bf*)g;
#endif
}
__device__ inline void wait_async0() {
#ifdef HAVE_ASYNC_LDS
#if __has_builtin(__builtin_amdgcn_s_wait_asynccnt)
    __builtin_amdgcn_s_wait_asynccnt(0);
#else
    asm volatile("s_wait_asynccnt 0" ::: "memory");
#endif
#endif
}

// =======================================================================
// small utility kernels
// =======================================================================
__global__ void init_loss_kernel(float* loss) { if (threadIdx.x == 0) loss[0] = 0.f; }

__global__ void cast_flat_kernel(const float* __restrict__ s, __bf16* __restrict__ d, long n) {
    long i = (long)blockIdx.x * blockDim.x + threadIdx.x;
    long stride = (long)gridDim.x * blockDim.x;
    for (; i < n; i += stride) d[i] = (__bf16)s[i];
}

// rows x dld bf16 destination; cols copied from src, tail [cols,dld) zero-filled
__global__ void cast_rows_kernel(const float* __restrict__ s, int sld,
                                 __bf16* __restrict__ d, int dld,
                                 long rows, int cols) {
    long total = rows * (long)dld;
    long i = (long)blockIdx.x * blockDim.x + threadIdx.x;
    long stride = (long)gridDim.x * blockDim.x;
    for (; i < total; i += stride) {
        long r = i / dld; int c = (int)(i - r * dld);
        d[i] = (c < cols) ? (__bf16)s[r * sld + c] : (__bf16)0.f;
    }
}

// W[K][N] f32 -> Wt[N][Kp] bf16 (transposed, K zero-padded to Kp)
__global__ void cast_transpose_kernel(const float* __restrict__ W,
                                      __bf16* __restrict__ Wt,
                                      int K, int Ncols, int Kp) {
    long total = (long)Ncols * Kp;
    long i = (long)blockIdx.x * blockDim.x + threadIdx.x;
    long stride = (long)gridDim.x * blockDim.x;
    for (; i < total; i += stride) {
        long n = i / Kp; int k = (int)(i - n * Kp);
        Wt[i] = (k < K) ? (__bf16)W[(long)k * Ncols + n] : (__bf16)0.f;
    }
}

// bbox features -> 128-wide linear; writes f32 + bf16 copies into the fused inputs
__global__ void bb_kernel(const float* __restrict__ ps, const float* __restrict__ pv,
                          const float* __restrict__ W_bb, const float* __restrict__ b_bb,
                          __bf16* __restrict__ s_bb_bf, __bf16* __restrict__ A_s_bf,
                          float* __restrict__ v_bb_f, __bf16* __restrict__ v_cat_bf) {
    int r = blockIdx.x, j = threadIdx.x;            // 128 threads = 128 out cols
    bool is_s = r < B_ * N_;
    int r2 = is_s ? r : r - B_ * N_;
    const float* p = is_s ? (ps + (long)r2 * 4) : (pv + (long)r2 * 4);
    float x0 = p[0], y0 = p[1], x1 = p[2], y1 = p[3];
    float sw = x1 - x0, sh = y1 - y0;
    float cx = x0 + 0.5f * sw, cy = y0 + 0.5f * sh;
    float area = sw * sh, shp = sw / (sh + 1e-14f);
    float f[10] = {x0, y0, x1, y1, sw, sh, cx, cy, area, shp};
    float acc = b_bb[j];
#pragma unroll
    for (int i = 0; i < 10; i++) acc += f[i] * W_bb[i * BBD_ + j];
    __bf16 hb = (__bf16)acc;
    if (is_s) {
        s_bb_bf[(long)r2 * BBD_ + j] = hb;
        A_s_bf[(long)r2 * DSIP_ + FSD_ + j] = hb;
    } else {
        v_bb_f[(long)r2 * BBD_ + j] = acc;
        v_cat_bf[(long)r2 * DV_ + FVD_ + j] = hb;
    }
}

// G_t[e][d] = sum_k sum_i W1[k,d,i]*W2[k,e,i]  (transposed 128x128 bilinear form)
__global__ void g_kernel(const float* __restrict__ W1, const float* __restrict__ W2,
                         __bf16* __restrict__ G_t) {
    __shared__ float w1row[INTER_];
    int d = blockIdx.x, e = threadIdx.x;
    float acc = 0.f;
    for (int k = 0; k < KH_; k++) {
        for (int i = threadIdx.x; i < INTER_; i += blockDim.x)
            w1row[i] = W1[((long)k * BBD_ + d) * INTER_ + i];
        __syncthreads();
        const float* w2 = W2 + ((long)k * BBD_ + e) * INTER_;
        float a = 0.f;
        for (int i = 0; i < INTER_; i++) a += w1row[i] * w2[i];
        acc += a;
        __syncthreads();
    }
    G_t[e * BBD_ + d] = (__bf16)acc;   // store transposed: [n][k]
}

// =======================================================================
// generic bf16 WMMA GEMM: out = epi(A[M,Kp] @ Wt[N,Kp]^T + bias)
//   A:  [Mrows][lda]  bf16, K zero-padded, lda%8==0
//   Wt: [Ncols][ldwt] bf16 (pre-transposed weights), K zero-padded
//   EPI 0: f32 out   EPI 1: bf16 out, scaled per (row/rpb, col)   EPI 2: relu f32
// 256 threads = 8 waves; 64x128 tile; wave = 16x64 = 4 WMMAs/K-step;
// double-buffered LDS; branch-free async-to-LDS staging (OOB rows clamped:
// garbage lands only in output rows/cols that the epilogue guards discard).
// =======================================================================
__device__ inline v16bf ld16(const __bf16* p0, const __bf16* p1) {
    v8bf x = *(const v8bf*)p0;
    v8bf y = *(const v8bf*)p1;
    v16bf r = {x[0], x[1], x[2], x[3], x[4], x[5], x[6], x[7],
               y[0], y[1], y[2], y[3], y[4], y[5], y[6], y[7]};
    return r;
}

#define ATILE (64 * 32)
#define BTILE (128 * 32)

template <int EPI>
__global__ void gemm_bf16_kernel(const __bf16* __restrict__ A, int lda,
                                 const __bf16* __restrict__ Wt, int ldwt,
                                 const float* __restrict__ bias,
                                 float* __restrict__ outF, __bf16* __restrict__ outB, int ldo,
                                 int Mrows, int Ncols, int Kp,
                                 const float* __restrict__ scale, int rpb) {
    __shared__ __attribute__((aligned(16))) __bf16 As[2 * ATILE];
    __shared__ __attribute__((aligned(16))) __bf16 Bs[2 * BTILE];
    const int tid = threadIdx.x;
    const int bm0 = blockIdx.y * 64;
    const int gn0 = blockIdx.x * 128;
    const int w = tid >> 5, lane = tid & 31;
    const int wm = w & 3;        // 16-row strip within 64
    const int wn = w >> 2;       // 64-col strip within 128

    v8f acc0 = {0.f, 0.f, 0.f, 0.f, 0.f, 0.f, 0.f, 0.f};
    v8f acc1 = acc0, acc2 = acc0, acc3 = acc0;

    // staging coordinates: A = 64x32 (1 chunk/thread), B = 128x32 (2 chunks/thread)
    const int a_m = tid >> 2, a_k0 = (tid & 3) * 8;
    const int b_n = tid >> 1, b_k0 = (tid & 1) * 16;
    const int a_gm = bm0 + a_m;
    const int b_gn = gn0 + b_n;
    // clamp OOB rows -> branch-free staging; garbage only reaches guarded stores
    const int a_row = a_gm < Mrows ? a_gm : Mrows - 1;
    const int b_row = b_gn < Ncols ? b_gn : Ncols - 1;
    const __bf16* Abase = A + (long)a_row * lda + a_k0;
    const __bf16* Bbase = Wt + (long)b_row * ldwt + b_k0;
    __bf16* la0 = As + a_m * 32 + a_k0;
    __bf16* lb0 = Bs + b_n * 32 + b_k0;

    // fragment coordinates (contiguous k = 16*(lane/16) + s for A and B)
    const int mrow = wm * 16 + (lane & 15);
    const int koff = (lane >> 4) * 8;
    const int kb   = (lane >> 4) * 16;

    auto stage = [&](int kk, int buf) {
        async_cp16(Abase + kk, la0 + buf * ATILE);
        async_cp16(Bbase + kk, lb0 + buf * BTILE);
        async_cp16(Bbase + kk + 8, lb0 + buf * BTILE + 8);
    };

    auto compute = [&](int buf) {
        const __bf16* Ab = As + buf * ATILE;
        const __bf16* Bb = Bs + buf * BTILE;
        v16bf af = ld16(Ab + mrow * 32 + kb, Ab + mrow * 32 + kb + 8);
        const int nbase = wn * 64 + (lane & 15);
        v16bf b0 = ld16(Bb + (nbase +  0) * 32 + kb, Bb + (nbase +  0) * 32 + kb + 8);
        v16bf b1 = ld16(Bb + (nbase + 16) * 32 + kb, Bb + (nbase + 16) * 32 + kb + 8);
        v16bf b2 = ld16(Bb + (nbase + 32) * 32 + kb, Bb + (nbase + 32) * 32 + kb + 8);
        v16bf b3 = ld16(Bb + (nbase + 48) * 32 + kb, Bb + (nbase + 48) * 32 + kb + 8);
        acc0 = __builtin_amdgcn_wmma_f32_16x16x32_bf16(false, af, false, b0, (short)0, acc0, false, false);
        acc1 = __builtin_amdgcn_wmma_f32_16x16x32_bf16(false, af, false, b1, (short)0, acc1, false, false);
        acc2 = __builtin_amdgcn_wmma_f32_16x16x32_bf16(false, af, false, b2, (short)0, acc2, false, false);
        acc3 = __builtin_amdgcn_wmma_f32_16x16x32_bf16(false, af, false, b3, (short)0, acc3, false, false);
    };

    stage(0, 0);
    int pb = 0;
    for (int kk = 0; kk < Kp; kk += 32, pb ^= 1) {
        wait_async0();
        __syncthreads();
        if (kk + 32 < Kp) stage(kk + 32, pb ^ 1);
        if (kk + 64 < Kp) {   // prefetch tile after next (global_prefetch_b8)
            __builtin_prefetch(Abase + kk + 64, 0, 1);
            __builtin_prefetch(Bbase + kk + 64, 0, 1);
        }
        compute(pb);
        __syncthreads();
    }

    // ---- epilogue: C layout VGPR r -> M = r + 8*(lane>=16), N = lane%16 ----
#pragma unroll
    for (int r = 0; r < 8; r++) {
        int m = wm * 16 + r + ((lane >> 4) << 3);
        int grow = bm0 + m;
        if (grow >= Mrows) continue;
#pragma unroll
        for (int c = 0; c < 4; c++) {
            int gc = gn0 + wn * 64 + c * 16 + (lane & 15);
            if (gc >= Ncols) continue;
            float a = (c == 0) ? acc0[r] : (c == 1) ? acc1[r] : (c == 2) ? acc2[r] : acc3[r];
            float x = a + (bias ? bias[gc] : 0.f);
            if (EPI == 0) {
                outF[(long)grow * ldo + gc] = x;
            } else if (EPI == 1) {
                outB[(long)grow * ldo + gc] =
                    (__bf16)(x * scale[(long)(grow / rpb) * Ncols + gc]);
            } else {
                outF[(long)grow * ldo + gc] = fmaxf(x, 0.f);
            }
        }
    }
}

// =======================================================================
// softmax over a row of `cols` (for g2 = softmax(l@W_l2+b))
// =======================================================================
__global__ void softmax_rows_kernel(float* __restrict__ x, int cols) {
    __shared__ float red[256];
    int tid = threadIdx.x;
    float* row = x + (long)blockIdx.x * cols;
    float mx = NEG_INF;
    for (int i = tid; i < cols; i += 256) mx = fmaxf(mx, row[i]);
    red[tid] = mx; __syncthreads();
    for (int s = 128; s > 0; s >>= 1) { if (tid < s) red[tid] = fmaxf(red[tid], red[tid + s]); __syncthreads(); }
    float m = red[0]; __syncthreads();
    float sum = 0.f;
    for (int i = tid; i < cols; i += 256) sum += expf(row[i] - m);
    red[tid] = sum; __syncthreads();
    for (int s = 128; s > 0; s >>= 1) { if (tid < s) red[tid] += red[tid + s]; __syncthreads(); }
    float denom = red[0]; __syncthreads();
    for (int i = tid; i < cols; i += 256) row[i] = expf(row[i] - m) / denom;
}

// =======================================================================
// adj row kernel: raw = (Tt_row . v_bb^T)/K -> softmax -> p^4 renorm -> mask
// =======================================================================
__device__ inline float bredmax128(float* red, float v) {
    int t = threadIdx.x;
    red[t] = v; __syncthreads();
    for (int s = 64; s > 0; s >>= 1) { if (t < s) red[t] = fmaxf(red[t], red[t + s]); __syncthreads(); }
    float r = red[0]; __syncthreads(); return r;
}
__device__ inline float bredsum128(float* red, float v) {
    int t = threadIdx.x;
    red[t] = v; __syncthreads();
    for (int s = 64; s > 0; s >>= 1) { if (t < s) red[t] += red[t + s]; __syncthreads(); }
    float r = red[0]; __syncthreads(); return r;
}

__global__ void adj_kernel(const float* __restrict__ Tt, const float* __restrict__ v_bb_f,
                           const int* __restrict__ mask_s,
                           float* __restrict__ adj_ws, float* __restrict__ adj_inf,
                           float* __restrict__ loss) {
    __shared__ float tl[BBD_];
    __shared__ float red[128];
    int row = blockIdx.x;                    // b*N + n
    int b = row / N_, n = row - b * N_;
    int tid = threadIdx.x;
    tl[tid] = Tt[(long)row * BBD_ + tid];
    __syncthreads();

    bool act = tid < M_;
    float val = 0.f;
    if (act) {
        const float* vb = v_bb_f + ((long)b * M_ + tid) * BBD_;
        float a = 0.f;
        for (int d = 0; d < BBD_; d++) a += tl[d] * vb[d];
        val = a * (1.0f / (float)KH_);
    }
    float rmax = bredmax128(red, act ? val : NEG_INF);
    float e = act ? expf(val - rmax) : 0.f;
    float s1 = bredsum128(red, e);
    float p = e / s1;
    float p2 = p * p, p4 = p2 * p2;
    float s4 = bredsum128(red, act ? p4 : 0.f);
    float adjv = act ? (p4 / s4) : 0.f;

    int msk = mask_s[b];
    bool token = n < msk;
    float adj_final = token ? adjv : 0.f;
    if (act) {
        adj_ws[(long)row * M_ + tid] = adj_final;
        adj_inf[(long)row * M_ + tid] = token ? adjv : NEG_INF;
    }
    float ss = bredsum128(red, adj_final * adj_final);
    if (tid == 0) atomicAdd(loss, -sqrtf(ss) / ((float)msk * 10.f));
}

// =======================================================================
// v_out[b,m,d] = eps * sum_n adj[b,n,m] * prepared_s[b,n,d] + v_ori[b,m,d]
// =======================================================================
__global__ void vout_kernel(const float* __restrict__ adj_ws, const float* __restrict__ prepared_s,
                            const float* __restrict__ v_ori, const float* __restrict__ eps_p,
                            float* __restrict__ v_out) {
    __shared__ float adj_l[N_ * M_];     // 5000 f32
    __shared__ float ps_l[N_ * 128];     // 6400 f32
    int b = blockIdx.x >> 4;
    int dt = blockIdx.x & 15;
    int tid = threadIdx.x;
    for (int i = tid; i < N_ * M_; i += 256) adj_l[i] = adj_ws[(long)b * N_ * M_ + i];
    for (int i = tid; i < N_ * 128; i += 256) {
        int n = i >> 7, dd = i & 127;
        ps_l[i] = prepared_s[((long)b * N_ + n) * FVD_ + dt * 128 + dd];
    }
    __syncthreads();
    float eps = eps_p[0];
    int dd = tid & 127, mh = tid >> 7;
    for (int it = 0; it < 50; it++) {
        int m = it * 2 + mh;
        float acc = 0.f;
        for (int n = 0; n < N_; n++) acc += adj_l[n * M_ + m] * ps_l[n * 128 + dd];
        long oi = ((long)b * M_ + m) * FVD_ + dt * 128 + dd;
        v_out[oi] = eps * acc + v_ori[oi];
    }
}

// =======================================================================
// s_out[b,n,:300] = mask*s ; s_out[b,n,300:] = mask * (adj_row . prepared_v)
// =======================================================================
__global__ void sout_kernel(const float* __restrict__ s, const float* __restrict__ adj_ws,
                            const float* __restrict__ prepared_v, const int* __restrict__ mask_s,
                            float* __restrict__ s_out) {
    __shared__ float adj_l[M_];
    int row = blockIdx.x;
    int b = row / N_, n = row - b * N_;
    int tid = threadIdx.x;
    if (tid < M_) adj_l[tid] = adj_ws[(long)row * M_ + tid];
    __syncthreads();
    bool token = n < mask_s[b];
    if (tid < FSD_) {
        s_out[(long)row * SOUT_ + tid] = token ? s[(long)row * FSD_ + tid] : 0.f;
        float acc = 0.f;
        for (int m = 0; m < M_; m++)
            acc += adj_l[m] * prepared_v[((long)b * M_ + m) * FSD_ + tid];
        s_out[(long)row * SOUT_ + FSD_ + tid] = token ? acc : 0.f;
    }
}

// =======================================================================
// host-side orchestration
// =======================================================================
extern "C" void kernel_launch(void* const* d_in, const int* in_sizes, int n_in,
                              void* d_out, int out_size, void* d_ws, size_t ws_size,
                              hipStream_t stream) {
    const float* l      = (const float*)d_in[0];
    const float* s      = (const float*)d_in[1];
    const float* ps     = (const float*)d_in[2];
    const int*   mask_s = (const int*)  d_in[3];
    const float* v_ori  = (const float*)d_in[4];
    const float* pv     = (const float*)d_in[5];
    const float* W_bb   = (const float*)d_in[7];
    const float* b_bb   = (const float*)d_in[8];
    const float* W1     = (const float*)d_in[9];
    const float* W2     = (const float*)d_in[10];
    const float* W_fs   = (const float*)d_in[11];
    const float* b_fs   = (const float*)d_in[12];
    const float* W_l2   = (const float*)d_in[13];
    const float* b_l2   = (const float*)d_in[14];
    const float* W_l3   = (const float*)d_in[15];
    const float* b_l3   = (const float*)d_in[16];
    const float* W_fv   = (const float*)d_in[17];
    const float* b_fv   = (const float*)d_in[18];
    const float* W_o1   = (const float*)d_in[19];
    const float* b_o1   = (const float*)d_in[20];
    const float* W_o2   = (const float*)d_in[21];
    const float* b_o2   = (const float*)d_in[22];
    const float* eps    = (const float*)d_in[23];

    float* out      = (float*)d_out;
    float* s_out    = out;                               // 256*50*600
    float* v_out    = out + 7680000L;                    // 256*100*2048
    float* adj_inf  = out + 7680000L + 52428800L;        // 256*50*100
    float* loss     = out + 61388800L;                   // 1

    char* wsp = (char*)d_ws;
    size_t off = 0;
    auto alloc = [&](size_t bytes) -> void* {
        void* p = wsp + off;
        off = (off + bytes + 255) & ~(size_t)255;
        return p;
    };
    float*  v_bb_f   = (float*) alloc((size_t)B_ * M_ * BBD_ * 4);
    __bf16* s_bb_bf  = (__bf16*)alloc((size_t)B_ * N_ * BBD_ * 2);
    float*  Tt_f     = (float*) alloc((size_t)B_ * N_ * BBD_ * 4);
    float*  g2       = (float*) alloc((size_t)B_ * DV_ * 4);
    float*  g3       = (float*) alloc((size_t)B_ * DV_ * 4);
    float*  adj_ws   = (float*) alloc((size_t)B_ * N_ * M_ * 4);
    float*  prep_s   = (float*) alloc((size_t)B_ * N_ * FVD_ * 4);
    float*  prep_v   = (float*) alloc((size_t)B_ * M_ * FSD_ * 4);
    __bf16* l_bf     = (__bf16*)alloc((size_t)B_ * LDIM_ * 2);
    __bf16* A_s_bf   = (__bf16*)alloc((size_t)B_ * N_ * DSIP_ * 2);
    __bf16* v_cat_bf = (__bf16*)alloc((size_t)B_ * M_ * DV_ * 2);
    __bf16* h_s_bf   = (__bf16*)alloc((size_t)B_ * N_ * DV_ * 2);
    __bf16* h_v_bf   = (__bf16*)alloc((size_t)B_ * M_ * DV_ * 2);
    __bf16* G_t      = (__bf16*)alloc((size_t)BBD_ * BBD_ * 2);
    __bf16* W_fs_t   = (__bf16*)alloc((size_t)DV_ * DSIP_ * 2);     // [2176][448]
    __bf16* W_l2_t   = (__bf16*)alloc((size_t)DV_ * LDIM_ * 2);     // [2176][2048]
    __bf16* W_l3_t   = (__bf16*)alloc((size_t)DV_ * LDIM_ * 2);
    __bf16* W_fv_t   = (__bf16*)alloc((size_t)DV_ * DV_ * 2);       // [2176][2176]
    __bf16* W_o1_t   = (__bf16*)alloc((size_t)FSD_ * DV_ * 2);      // [300][2176]
    __bf16* W_o2_t   = (__bf16*)alloc((size_t)FVD_ * DV_ * 2);      // [2048][2176]
    (void)ws_size; (void)in_sizes; (void)n_in; (void)out_size;

    init_loss_kernel<<<1, 1, 0, stream>>>(loss);

    // weight transposes (to [N][Kp] bf16) + activation downconversions
    cast_transpose_kernel<<<2048, 256, 0, stream>>>(W_fs, W_fs_t, DSI_, DV_, DSIP_);
    cast_transpose_kernel<<<2048, 256, 0, stream>>>(W_l2, W_l2_t, LDIM_, DV_, LDIM_);
    cast_transpose_kernel<<<2048, 256, 0, stream>>>(W_l3, W_l3_t, LDIM_, DV_, LDIM_);
    cast_transpose_kernel<<<2048, 256, 0, stream>>>(W_fv, W_fv_t, DV_, DV_, DV_);
    cast_transpose_kernel<<<2048, 256, 0, stream>>>(W_o1, W_o1_t, DV_, FSD_, DV_);
    cast_transpose_kernel<<<2048, 256, 0, stream>>>(W_o2, W_o2_t, DV_, FVD_, DV_);
    cast_flat_kernel<<<2048, 256, 0, stream>>>(l, l_bf, (long)B_ * LDIM_);
    cast_rows_kernel<<<2048, 256, 0, stream>>>(s, FSD_, A_s_bf, DSIP_, (long)B_ * N_, FSD_);
    cast_rows_kernel<<<2048, 256, 0, stream>>>(v_ori, FVD_, v_cat_bf, DV_, (long)B_ * M_, FVD_);

    // bbox features + 10->128 linear (fills concat tails; after cast_rows zero-pad)
    bb_kernel<<<B_ * N_ + B_ * M_, 128, 0, stream>>>(ps, pv, W_bb, b_bb,
                                                     s_bb_bf, A_s_bf, v_bb_f, v_cat_bf);

    // G = sum_k W1_k W2_k^T (stored transposed)
    g_kernel<<<128, 128, 0, stream>>>(W1, W2, G_t);

    // g3 = l @ W_l3 + b_l3 ; g2 = softmax(l @ W_l2 + b_l2)
    gemm_bf16_kernel<0><<<dim3(17, 4), 256, 0, stream>>>(
        l_bf, LDIM_, W_l3_t, LDIM_, b_l3, g3, nullptr, DV_, B_, DV_, LDIM_, nullptr, 1);
    gemm_bf16_kernel<0><<<dim3(17, 4), 256, 0, stream>>>(
        l_bf, LDIM_, W_l2_t, LDIM_, b_l2, g2, nullptr, DV_, B_, DV_, LDIM_, nullptr, 1);
    softmax_rows_kernel<<<B_, 256, 0, stream>>>(g2, DV_);

    // Tt = s_bb @ G
    gemm_bf16_kernel<0><<<dim3(1, 200), 256, 0, stream>>>(
        s_bb_bf, BBD_, G_t, BBD_, nullptr, Tt_f, nullptr, BBD_,
        B_ * N_, BBD_, BBD_, nullptr, 1);

    // h_s = ([s|s_bb] @ W_fs + b_fs) * g3   (bf16 epilogue, scale fused)
    gemm_bf16_kernel<1><<<dim3(17, 200), 256, 0, stream>>>(
        A_s_bf, DSIP_, W_fs_t, DSIP_, b_fs, nullptr, h_s_bf, DV_,
        B_ * N_, DV_, DSIP_, g3, N_);
    // h_v = ([v_ori|v_bb] @ W_fv + b_fv) * softmax(g2)
    gemm_bf16_kernel<1><<<dim3(17, 400), 256, 0, stream>>>(
        v_cat_bf, DV_, W_fv_t, DV_, b_fv, nullptr, h_v_bf, DV_,
        B_ * M_, DV_, DV_, g2, M_);

    // prepared_s = relu(h_s @ W_o2 + b_o2) ; prepared_v = relu(h_v @ W_o1 + b_o1)
    gemm_bf16_kernel<2><<<dim3(16, 200), 256, 0, stream>>>(
        h_s_bf, DV_, W_o2_t, DV_, b_o2, prep_s, nullptr, FVD_,
        B_ * N_, FVD_, DV_, nullptr, 1);
    gemm_bf16_kernel<2><<<dim3(3, 400), 256, 0, stream>>>(
        h_v_bf, DV_, W_o1_t, DV_, b_o1, prep_v, nullptr, FSD_,
        B_ * M_, FSD_, DV_, nullptr, 1);

    // adjacency: softmax + p^4 renorm + mask + adj_inf + loss
    adj_kernel<<<B_ * N_, 128, 0, stream>>>(Tt_f, v_bb_f, mask_s, adj_ws, adj_inf, loss);

    // v_out and s_out einsums
    vout_kernel<<<B_ * 16, 256, 0, stream>>>(adj_ws, prep_s, v_ori, eps, v_out);
    sout_kernel<<<B_ * N_, 320, 0, stream>>>(s, adj_ws, prep_v, mask_s, s_out);
}
